// RotorQuantLayer_19250043421384
// MI455X (gfx1250) — compile-verified, matrix operand order checked by reading, not run on hardware
//
#include <hip/hip_runtime.h>

// RotorQuantLayer for MI455X (gfx1250, wave32):
//   y = x @ W^T + b          -> bf16 hi/lo split GEMM via v_wmma_f32_16x16x32_bf16
//   z = FWHT(y*flips)/64     -> per-row FWHT in registers + LDS
//   q = cent[searchsorted(bp, z)]
//   out = FWHT(q)/64 * flips
// Memory-bound (~1.14 GB HBM traffic @ 23.3 TB/s); GEMM compute is negligible.

typedef __attribute__((ext_vector_type(16))) __bf16 v16bf;
typedef __attribute__((ext_vector_type(8)))  float  v8f;

#define D_DIM 4096
#define BM 128
#define BN 128
#define BK 32

// ---------------- fp32 -> bf16 hi/lo split helpers ----------------

__device__ __forceinline__ unsigned short f2bf_rn(float f) {
  unsigned u = __float_as_uint(f);
  u += 0x7FFFu + ((u >> 16) & 1u);   // round-to-nearest-even
  return (unsigned short)(u >> 16);
}
__device__ __forceinline__ float bf2f(unsigned short h) {
  return __uint_as_float(((unsigned)h) << 16);
}
__device__ __forceinline__ void pack_hilo(float f, unsigned short& hi, unsigned short& lo) {
  hi = f2bf_rn(f);
  lo = f2bf_rn(f - bf2f(hi));
}

__device__ __forceinline__ void store_hilo4(unsigned short* hiBase, unsigned short* loBase,
                                            float4 v) {
  unsigned short h0, h1, h2, h3, l0, l1, l2, l3;
  pack_hilo(v.x, h0, l0);
  pack_hilo(v.y, h1, l1);
  pack_hilo(v.z, h2, l2);
  pack_hilo(v.w, h3, l3);
  uint2 uh, ul;
  uh.x = (unsigned)h0 | ((unsigned)h1 << 16);
  uh.y = (unsigned)h2 | ((unsigned)h3 << 16);
  ul.x = (unsigned)l0 | ((unsigned)l1 << 16);
  ul.y = (unsigned)l2 | ((unsigned)l3 << 16);
  *(uint2*)hiBase = uh;   // ds_store_b64
  *(uint2*)loBase = ul;
}

// ---------------- Phase 1: y = x @ W^T + b (bf16x3 WMMA) ----------------
// Block tile 128x128, K-step 32, 8 waves (2 along M x 4 along N),
// each wave computes 4x2 = 8 16x16 f32 accumulators, 3 WMMAs each per K-step.

__global__ __launch_bounds__(256)
void gemm_bf16x3_kernel(const float* __restrict__ X,
                        const float* __restrict__ Wm,
                        const float* __restrict__ bias,
                        float* __restrict__ Y) {
  __shared__ __align__(32) unsigned short As_hi[BM * BK];
  __shared__ __align__(32) unsigned short As_lo[BM * BK];
  __shared__ __align__(32) unsigned short Bs_hi[BN * BK];
  __shared__ __align__(32) unsigned short Bs_lo[BN * BK];

  const int tid   = threadIdx.x;
  const int lane  = tid & 31;
  const int wave  = tid >> 5;
  const int waveM = wave & 1;   // 2 waves along M (64 rows each)
  const int waveN = wave >> 1;  // 4 waves along N (32 cols each)

  const size_t mBlock = (size_t)blockIdx.y * BM;
  const size_t nBlock = (size_t)blockIdx.x * BN;

  // WMMA fragment addressing (wave32): lanes 0-15 carry row/col index,
  // lanes 16-31 carry the upper K half (elements 16..31).
  const int laneRow = lane & 15;
  const int kHalf   = (lane >> 4) << 4;   // 0 or 16 bf16 elements

  v8f acc[4][2] = {};

  for (int k0 = 0; k0 < D_DIM; k0 += BK) {
    // ---- global -> LDS, fp32 -> bf16 hi/lo (each thread: 4x float4 per tile)
#pragma unroll
    for (int p = 0; p < 4; ++p) {
      const int flat = p * 1024 + tid * 4;        // 0..4095
      const int rr   = flat >> 5;                 // row in tile
      const int cc   = flat & 31;                 // col in tile
      const float4 xa = *(const float4*)(X  + (mBlock + rr) * D_DIM + k0 + cc);
      const float4 wb = *(const float4*)(Wm + (nBlock + rr) * D_DIM + k0 + cc);
      store_hilo4(&As_hi[flat], &As_lo[flat], xa);
      store_hilo4(&Bs_hi[flat], &Bs_lo[flat], wb);
    }
    __syncthreads();

    // ---- B fragments for this wave's two N tiles (rows of W = cols of W^T)
    v16bf bhi[2], blo[2];
#pragma unroll
    for (int j = 0; j < 2; ++j) {
      const int nrow = waveN * 32 + j * 16 + laneRow;
      bhi[j] = *(const v16bf*)&Bs_hi[nrow * BK + kHalf];   // 32B -> 2x ds_load_b128
      blo[j] = *(const v16bf*)&Bs_lo[nrow * BK + kHalf];
    }

    // ---- A fragments + 3-term bf16 WMMA accumulate
#pragma unroll
    for (int i = 0; i < 4; ++i) {
      const int mrow = waveM * 64 + i * 16 + laneRow;
      const v16bf ahi = *(const v16bf*)&As_hi[mrow * BK + kHalf];
      const v16bf alo = *(const v16bf*)&As_lo[mrow * BK + kHalf];
#pragma unroll
      for (int j = 0; j < 2; ++j) {
        acc[i][j] = __builtin_amdgcn_wmma_f32_16x16x32_bf16(
            false, ahi, false, bhi[j], (short)0, acc[i][j], false, false);
        acc[i][j] = __builtin_amdgcn_wmma_f32_16x16x32_bf16(
            false, ahi, false, blo[j], (short)0, acc[i][j], false, false);
        acc[i][j] = __builtin_amdgcn_wmma_f32_16x16x32_bf16(
            false, alo, false, bhi[j], (short)0, acc[i][j], false, false);
      }
    }
    __syncthreads();
  }

  // ---- epilogue: C layout (16x16 f32): VGPR e holds M=e (lanes 0-15) / M=e+8 (lanes 16-31)
  const int rHalf = (lane >> 4) << 3;   // +8 rows for upper half-wave
#pragma unroll
  for (int j = 0; j < 2; ++j) {
    const size_t col = nBlock + (size_t)(waveN * 32 + j * 16 + laneRow);
    const float  bv  = bias[col];
#pragma unroll
    for (int i = 0; i < 4; ++i) {
      const size_t rowBase = mBlock + (size_t)(waveM * 64 + i * 16 + rHalf);
#pragma unroll
      for (int e = 0; e < 8; ++e) {
        Y[(rowBase + e) * D_DIM + col] = acc[i][j][e] + bv;
      }
    }
  }
}

// ---------------- Phase 2: per-row FWHT -> quantize -> FWHT (in-place) ----------------
// 256 threads per 4096-float row; each thread owns 16 consecutive elements.
// Stages h=1..8 in registers; stages h=16..2048 as chunk exchanges through LDS.

__device__ __forceinline__ void fwht4096_block(float r[16], float* lds, int t) {
  // intra-chunk stages (h = 1,2,4,8)
#pragma unroll
  for (int h = 1; h <= 8; h <<= 1) {
#pragma unroll
    for (int j = 0; j < 16; ++j) {
      if ((j & h) == 0) {
        const float a = r[j];
        const float b = r[j + h];
        r[j]     = a + b;
        r[j + h] = a - b;
      }
    }
  }
  // inter-chunk stages (chunk distance s = 1..128 over 256 chunks)
#pragma unroll
  for (int s = 1; s <= 128; s <<= 1) {
    __syncthreads();
    float4* mine = (float4*)&lds[t * 16];
    mine[0] = make_float4(r[0],  r[1],  r[2],  r[3]);
    mine[1] = make_float4(r[4],  r[5],  r[6],  r[7]);
    mine[2] = make_float4(r[8],  r[9],  r[10], r[11]);
    mine[3] = make_float4(r[12], r[13], r[14], r[15]);
    __syncthreads();
    const float4* part = (const float4*)&lds[(t ^ s) * 16];
    const float4 q0 = part[0], q1 = part[1], q2 = part[2], q3 = part[3];
    float p[16] = {q0.x, q0.y, q0.z, q0.w, q1.x, q1.y, q1.z, q1.w,
                   q2.x, q2.y, q2.z, q2.w, q3.x, q3.y, q3.z, q3.w};
    if (t & s) {
#pragma unroll
      for (int j = 0; j < 16; ++j) r[j] = p[j] - r[j];
    } else {
#pragma unroll
      for (int j = 0; j < 16; ++j) r[j] = r[j] + p[j];
    }
  }
}

__global__ __launch_bounds__(256)
void fwht_quant_kernel(float* __restrict__ y,
                       const float* __restrict__ flips,
                       const float* __restrict__ bp,
                       const float* __restrict__ cent) {
  __shared__ __align__(16) float lds[D_DIM];
  __shared__ float s_bp[16];
  __shared__ float s_cent[16];

  const int t = threadIdx.x;
  if (t < 15) s_bp[t]   = bp[t];
  if (t < 16) s_cent[t] = cent[t];

  float* rowp = y + (size_t)blockIdx.x * D_DIM;

  float r[16], fl[16];
#pragma unroll
  for (int p = 0; p < 4; ++p) {
    const float4 v = *(const float4*)(rowp  + t * 16 + p * 4);
    const float4 f = *(const float4*)(flips + t * 16 + p * 4);
    fl[p * 4 + 0] = f.x; fl[p * 4 + 1] = f.y; fl[p * 4 + 2] = f.z; fl[p * 4 + 3] = f.w;
    r[p * 4 + 0] = v.x * f.x;
    r[p * 4 + 1] = v.y * f.y;
    r[p * 4 + 2] = v.z * f.z;
    r[p * 4 + 3] = v.w * f.w;
  }
  __syncthreads();   // s_bp / s_cent visible

  fwht4096_block(r, lds, t);

  // z = fwht(y*flips)/64 ; idx = #{bp_i < z} (searchsorted side='left') ; q = cent[idx]
  const float scale = 0.015625f;   // 1/sqrt(4096)
  float bpr[15];
#pragma unroll
  for (int i = 0; i < 15; ++i) bpr[i] = s_bp[i];
#pragma unroll
  for (int j = 0; j < 16; ++j) {
    const float z = r[j] * scale;
    int idx = 0;
#pragma unroll
    for (int i = 0; i < 15; ++i) idx += (bpr[i] < z) ? 1 : 0;
    r[j] = s_cent[idx];
  }

  fwht4096_block(r, lds, t);

#pragma unroll
  for (int p = 0; p < 4; ++p) {
    float4 o;
    o.x = r[p * 4 + 0] * scale * fl[p * 4 + 0];
    o.y = r[p * 4 + 1] * scale * fl[p * 4 + 1];
    o.z = r[p * 4 + 2] * scale * fl[p * 4 + 2];
    o.w = r[p * 4 + 3] * scale * fl[p * 4 + 3];
    *(float4*)(rowp + t * 16 + p * 4) = o;
  }
}

// ---------------- launch ----------------

extern "C" void kernel_launch(void* const* d_in, const int* in_sizes, int n_in,
                              void* d_out, int out_size, void* d_ws, size_t ws_size,
                              hipStream_t stream) {
  (void)n_in; (void)out_size; (void)d_ws; (void)ws_size;
  const float* x     = (const float*)d_in[0];
  const float* W     = (const float*)d_in[1];
  const float* bias  = (const float*)d_in[2];
  const float* flips = (const float*)d_in[3];
  const float* bp    = (const float*)d_in[4];
  const float* cent  = (const float*)d_in[5];
  float* out = (float*)d_out;

  const int M = in_sizes[0] / D_DIM;   // 8*2048 = 16384 rows

  // Phase 1: GEMM writes y straight into d_out (same size as final output).
  // blockIdx.x over N (fastest) keeps the x row-strip L2-resident; W fits in L2.
  dim3 gGemm(D_DIM / BN, M / BM);
  gemm_bf16x3_kernel<<<gGemm, 256, 0, stream>>>(x, W, bias, out);

  // Phase 2: in-place per-row FWHT -> quantize -> FWHT (row-exclusive blocks).
  fwht_quant_kernel<<<dim3(M), 256, 0, stream>>>(out, flips, bp, cent);
}
